// Decoder_41300405518511
// MI455X (gfx1250) — compile-verified, matrix operand order checked by reading, not run on hardware
//
#include <hip/hip_runtime.h>
#include <hip/hip_bf16.h>
#include <math.h>

// ---- WMMA vector types (gfx1250, wave32) ----
typedef __attribute__((ext_vector_type(16))) __bf16 v16bf;
typedef __attribute__((ext_vector_type(8)))  float  v8f;

// Problem constants (fixed by reference setup_inputs)
#define NB     8
#define NT     16
#define NIMG   (NB * NT)     // 128
#define NCH    128           // channels C == h_dim
#define HG     64
#define WG_    64
#define NP     2048          // points per image
#define FMAP   128
#define INDIM  (NCH + 2 * FMAP)   // 384
#define HDIM   128
#define PTS    64            // points per workgroup (4 M-tiles)
#define TWO_PI 6.28318530717958647692f
#define MIN_STD 0.01f

// ---------------------------------------------------------------------------
// Pre-kernel: W0 (384x128 f32, row-major K x N) -> w0t bf16 [N=128][K=384]
//             W1 (128x128 f32)                  -> w1t bf16 [N=128][K=128]
// ---------------------------------------------------------------------------
__global__ __launch_bounds__(256) void convert_weights_kernel(
    const float* __restrict__ W0, const float* __restrict__ W1,
    unsigned short* __restrict__ w0t_u, unsigned short* __restrict__ w1t_u)
{
    __bf16* w0t = reinterpret_cast<__bf16*>(w0t_u);
    __bf16* w1t = reinterpret_cast<__bf16*>(w1t_u);
    int i = blockIdx.x * 256 + threadIdx.x;
    const int n0 = INDIM * HDIM;           // 49152
    const int n1 = HDIM * HDIM;            // 16384
    if (i < n0) {
        int k = i / HDIM, n = i % HDIM;
        w0t[n * INDIM + k] = (__bf16)W0[i];
    } else if (i < n0 + n1) {
        int j = i - n0;
        int k = j / HDIM, n = j % HDIM;
        w1t[n * HDIM + k] = (__bf16)W1[j];
    }
}

// ---------------------------------------------------------------------------
// Main fused kernel: gather+fourier -> LDS(bf16) -> WMMA MLP -> head
// Grid: (NP/PTS, NIMG), Block: 256 threads (8 wave32)
// Wave w owns N-tile nb=w and iterates all 4 M-tiles: B-fragment fetched
// once per k-step, reused by 4 WMMAs (4x less global weight traffic).
// ---------------------------------------------------------------------------
__global__ __launch_bounds__(256) void decoder_kernel(
    const float* __restrict__ h_grid,   // (NIMG, 128, 64, 64)
    const float* __restrict__ xs,       // (NIMG, P)
    const float* __restrict__ ys,       // (NIMG, P)
    const float* __restrict__ Bm,       // (2, 128)
    const float* __restrict__ b0,       // (128,)
    const float* __restrict__ b1,       // (128,)
    const float* __restrict__ W2,       // (128, 2) row-major
    const float* __restrict__ b2,       // (2,)
    const unsigned short* __restrict__ w0t_u,  // bf16 [128][384]
    const unsigned short* __restrict__ w1t_u,  // bf16 [128][128]
    float* __restrict__ out)            // (NIMG, P, 2)
{
    const __bf16* w0t = reinterpret_cast<const __bf16*>(w0t_u);
    const __bf16* w1t = reinterpret_cast<const __bf16*>(w1t_u);

    __shared__ alignas(32) __bf16 xbuf[PTS * INDIM];  // 48 KB; reused as h2 later
    __shared__ alignas(32) __bf16 h1[PTS * HDIM];     // 16 KB

    const int img   = blockIdx.y;
    const int pbase = blockIdx.x * PTS;
    const int tid   = threadIdx.x;

    // ---------------- Phase 1: features (4 threads per point) --------------
    {
        const int p   = tid >> 2;        // 0..63
        const int sub = tid & 3;         // 0..3 -> 32 channels / 32 freqs each
        const int gp  = img * NP + pbase + p;
        const float cx = xs[gp];
        const float cy = ys[gp];

        // align_corners bilinear; g = 2c-1 maps back to c*(W-1)
        const float fx = cx * (float)(WG_ - 1);
        const float fy = cy * (float)(HG - 1);
        float x0f = floorf(fx); x0f = fminf(fmaxf(x0f, 0.0f), (float)(WG_ - 1));
        float y0f = floorf(fy); y0f = fminf(fmaxf(y0f, 0.0f), (float)(HG - 1));
        const int x0 = (int)x0f;
        const int y0 = (int)y0f;
        const int x1 = min(x0 + 1, WG_ - 1);
        const int y1 = min(y0 + 1, HG - 1);
        const float wx = fx - x0f;
        const float wy = fy - y0f;

        const float* base = h_grid + (size_t)img * NCH * HG * WG_;
        const int i00 = y0 * WG_ + x0;
        const int i01 = y0 * WG_ + x1;
        const int i10 = y1 * WG_ + x0;
        const int i11 = y1 * WG_ + x1;

        __bf16* xrow = &xbuf[p * INDIM];
        const int cbeg = sub * 32;
        #pragma unroll 4
        for (int c = cbeg; c < cbeg + 32; ++c) {
            const float* pl = base + (size_t)c * (HG * WG_);
            const float v00 = pl[i00];
            const float v01 = pl[i01];
            const float v10 = pl[i10];
            const float v11 = pl[i11];
            const float top = v00 + (v01 - v00) * wx;
            const float bot = v10 + (v11 - v10) * wx;
            xrow[c] = (__bf16)(top + (bot - top) * wy);
        }
        const int fbeg = sub * 32;
        #pragma unroll 4
        for (int f = fbeg; f < fbeg + 32; ++f) {
            const float pr = TWO_PI * (cx * Bm[f] + cy * Bm[FMAP + f]);
            float sv, cv;
            __sincosf(pr, &sv, &cv);
            xrow[NCH + f]        = (__bf16)sv;
            xrow[NCH + FMAP + f] = (__bf16)cv;
        }
    }
    __syncthreads();

    const int wave  = tid >> 5;          // 0..7 == N-tile index
    const int lane  = tid & 31;
    const int lrow  = lane & 15;         // row/col within 16-tile
    const int khalf = lane >> 4;         // 0: K 0-15, 1: K 16-31
    const int col   = wave * 16 + lrow;  // output column owned by this lane

    // ---------------- Phase 2: GEMM0  (64x384) @ (384x128) -----------------
    {
        v8f acc0 = {}, acc1 = {}, acc2 = {}, acc3 = {};
        const __bf16* arow = &xbuf[lrow * INDIM + khalf * 16];
        const __bf16* brow = &w0t[col * INDIM + khalf * 16];
        #pragma unroll
        for (int k = 0; k < INDIM / 32; ++k) {   // 12 steps
            v16bf b  = *(const v16bf*)(brow + k * 32);
            v16bf a0 = *(const v16bf*)(arow + k * 32);
            v16bf a1 = *(const v16bf*)(arow + k * 32 + 16 * INDIM);
            v16bf a2 = *(const v16bf*)(arow + k * 32 + 32 * INDIM);
            v16bf a3 = *(const v16bf*)(arow + k * 32 + 48 * INDIM);
            acc0 = __builtin_amdgcn_wmma_f32_16x16x32_bf16(false, a0, false, b, (short)0, acc0, false, false);
            acc1 = __builtin_amdgcn_wmma_f32_16x16x32_bf16(false, a1, false, b, (short)0, acc1, false, false);
            acc2 = __builtin_amdgcn_wmma_f32_16x16x32_bf16(false, a2, false, b, (short)0, acc2, false, false);
            acc3 = __builtin_amdgcn_wmma_f32_16x16x32_bf16(false, a3, false, b, (short)0, acc3, false, false);
        }
        const float bias = b0[col];
        v8f accs[4] = {acc0, acc1, acc2, acc3};
        #pragma unroll
        for (int m = 0; m < 4; ++m) {
            #pragma unroll
            for (int r = 0; r < 8; ++r) {
                const int row = m * 16 + r + khalf * 8;   // C/D: lanes16-31 -> M+8
                float v = accs[m][r] + bias;
                h1[row * HDIM + col] = (__bf16)(v > 0.0f ? v : 0.0f);
            }
        }
    }
    __syncthreads();

    // ---------------- Phase 3: GEMM1  (64x128) @ (128x128) -----------------
    __bf16* h2 = xbuf;   // reuse feature buffer (free after GEMM0)
    {
        v8f acc0 = {}, acc1 = {}, acc2 = {}, acc3 = {};
        const __bf16* arow = &h1[lrow * HDIM + khalf * 16];
        const __bf16* brow = &w1t[col * HDIM + khalf * 16];
        #pragma unroll
        for (int k = 0; k < HDIM / 32; ++k) {    // 4 steps
            v16bf b  = *(const v16bf*)(brow + k * 32);
            v16bf a0 = *(const v16bf*)(arow + k * 32);
            v16bf a1 = *(const v16bf*)(arow + k * 32 + 16 * HDIM);
            v16bf a2 = *(const v16bf*)(arow + k * 32 + 32 * HDIM);
            v16bf a3 = *(const v16bf*)(arow + k * 32 + 48 * HDIM);
            acc0 = __builtin_amdgcn_wmma_f32_16x16x32_bf16(false, a0, false, b, (short)0, acc0, false, false);
            acc1 = __builtin_amdgcn_wmma_f32_16x16x32_bf16(false, a1, false, b, (short)0, acc1, false, false);
            acc2 = __builtin_amdgcn_wmma_f32_16x16x32_bf16(false, a2, false, b, (short)0, acc2, false, false);
            acc3 = __builtin_amdgcn_wmma_f32_16x16x32_bf16(false, a3, false, b, (short)0, acc3, false, false);
        }
        const float bias = b1[col];
        v8f accs[4] = {acc0, acc1, acc2, acc3};
        #pragma unroll
        for (int m = 0; m < 4; ++m) {
            #pragma unroll
            for (int r = 0; r < 8; ++r) {
                const int row = m * 16 + r + khalf * 8;
                float v = accs[m][r] + bias;
                h2[row * HDIM + col] = (__bf16)(v > 0.0f ? v : 0.0f);
            }
        }
    }
    __syncthreads();

    // ---------------- Phase 4: head (128 -> 2) + softplus -------------------
    if (tid < PTS * 2) {
        const int p = tid >> 1;
        const int o = tid & 1;
        float acc = b2[o];
        const __bf16* hrow = &h2[p * HDIM];
        #pragma unroll 8
        for (int c = 0; c < HDIM; ++c) {
            acc += (float)hrow[c] * W2[c * 2 + o];
        }
        float res;
        if (o == 0) {
            res = acc;                                 // mu
        } else {
            const float sp = (acc > 20.0f) ? acc : log1pf(__expf(acc));
            res = sp + MIN_STD;                        // sigma
        }
        out[(size_t)(img * NP + pbase + p) * 2 + o] = res;
    }
}

// ---------------------------------------------------------------------------
extern "C" void kernel_launch(void* const* d_in, const int* in_sizes, int n_in,
                              void* d_out, int out_size, void* d_ws, size_t ws_size,
                              hipStream_t stream) {
    const float* h_grid = (const float*)d_in[0];
    const float* xs     = (const float*)d_in[1];
    const float* ys     = (const float*)d_in[2];
    const float* Bm     = (const float*)d_in[3];
    const float* W0     = (const float*)d_in[4];
    const float* b0     = (const float*)d_in[5];
    const float* W1     = (const float*)d_in[6];
    const float* b1     = (const float*)d_in[7];
    const float* W2     = (const float*)d_in[8];
    const float* b2     = (const float*)d_in[9];

    unsigned short* w0t = (unsigned short*)d_ws;                 // 49152 bf16
    unsigned short* w1t = w0t + (INDIM * HDIM);                  // 16384 bf16

    convert_weights_kernel<<<(INDIM * HDIM + HDIM * HDIM + 255) / 256, 256, 0, stream>>>(
        W0, W1, w0t, w1t);

    dim3 grid(NP / PTS, NIMG);   // (32, 128)
    decoder_kernel<<<grid, 256, 0, stream>>>(
        h_grid, xs, ys, Bm, b0, b1, W2, b2, w0t, w1t, (float*)d_out);
}
